// EGNNQM9Model_56307021251053
// MI455X (gfx1250) — compile-verified, matrix operand order checked by reading, not run on hardware
//
#include <hip/hip_runtime.h>

// ---------------------------------------------------------------------------
// Problem constants (match reference)
// ---------------------------------------------------------------------------
#define BATCH   1024
#define NA      29
#define FDIM    64
#define DEPTH   4
#define KNN     8
#define NODES   (BATCH * NA)        // 29696  (divisible by 16)
#define EDGES   (NODES * KNN)       // 237568 (divisible by 16)
#define EDGE_IN 129
#define EHID    258
#define MDIM    16
#define NIN     80
#define NHID    128
#define LN_EPS  1e-5f
#define COORS_EPS 1e-8f
#define CLAMP_W 2.0f

// Padded WMMA GEMM geometry (K steps of 32, N tiles of 16)
#define E1_KS 5    // 129 -> 160
#define E1_NT 17   // 258 -> 272
#define E2_KS 9    // 258 -> 288 (272 used + zero tail)
#define C1_NT 4    // 16 -> 32 K-pad, N = 64
#define N1_KS 3    // 80 -> 96
#define N1_NT 8    // N = 128
#define N2_KS 4    // K = 128
#define N2_NT 4    // N = 64

#define PE1_HALFS (E1_KS * E1_NT * 512)  // 43520 per layer
#define PE2_HALFS (E2_KS * 1     * 512)  //  4608
#define PC1_HALFS (1     * C1_NT * 512)  //  2048
#define PN1_HALFS (N1_KS * N1_NT * 512)  // 12288
#define PN2_HALFS (N2_KS * N2_NT * 512)  //  8192

typedef __attribute__((ext_vector_type(16))) _Float16 v16h;
typedef __attribute__((ext_vector_type(8)))  _Float16 v8h;
typedef __attribute__((ext_vector_type(8)))  float    v8f;

// ---------------------------------------------------------------------------
// WMMA helpers (CDNA5 16x16x32 f16 -> f32)
// ---------------------------------------------------------------------------
__device__ __forceinline__ v8f wmma16(v16h a, v16h b, v8f c) {
    return __builtin_amdgcn_wmma_f32_16x16x32_f16(
        /*neg_a=*/false, a, /*neg_b=*/false, b,
        /*c_mod=*/(short)0, c, /*reuse_a=*/false, /*reuse_b=*/false);
}

// K index held by (vgpr pair r, lane-half h): ISA 16-bit A/B striping
__device__ __forceinline__ int kmap(int r, int h) {
    return (r < 4 ? 2 * r : 16 + 2 * (r - 4)) + 8 * h;
}

// Load a 16(M)x32(K) f16 A-fragment from an LDS row-major tile.
// Per the ISA striping, lane-half h needs K = [kOff+8h, +8) in VGPRs 0..3 and
// K = [kOff+16+8h, +8) in VGPRs 4..7  ->  exactly two aligned ds_load_b128.
// Rows are 16B-aligned for all ldk used here (32/96/128/160/288 halfs).
__device__ __forceinline__ v16h load_a_frag(const _Float16* X, int ldk, int kOff, int lane) {
    int idx = lane & 15, h = lane >> 4;
    const _Float16* row = X + idx * ldk + kOff + 8 * h;
    v8h lo = *(const v8h*)(row);        // K = kOff+8h   .. +7
    v8h hi = *(const v8h*)(row + 16);   // K = kOff+16+8h.. +7
    v16h a;
#pragma unroll
    for (int i = 0; i < 8; ++i) { a[i] = lo[i]; a[8 + i] = hi[i]; }
    return a;
}

__device__ __forceinline__ v8f zero8() {
    v8f z;
#pragma unroll
    for (int i = 0; i < 8; ++i) z[i] = 0.0f;
    return z;
}

// Fast SiLU: result feeds f16 WMMA operands, so v_rcp_f32 precision suffices.
__device__ __forceinline__ float silu(float x) {
    return x * __builtin_amdgcn_rcpf(1.0f + __expf(-x));
}

// ---------------------------------------------------------------------------
// Pack f32 weight matrix W[Kd][Nd] into WMMA B-fragment layout:
// dst[(ks*ntiles + nt)*512 + lane*16 + e]; lane loads its fragment contiguously
// ---------------------------------------------------------------------------
__global__ void egnn_pack_kernel(const float* __restrict__ W, int Kd, int Nd,
                                 int ksteps, int ntiles, _Float16* __restrict__ dst) {
    int t = blockIdx.x * blockDim.x + threadIdx.x;
    int total = ksteps * ntiles * 512;
    if (t >= total) return;
    int e    = t & 15;
    int lane = (t >> 4) & 31;
    int tile = t >> 9;
    int nt = tile % ntiles, ks = tile / ntiles;
    int h = lane >> 4, nl = lane & 15;
    int r = e >> 1;
    int k = ks * 32 + kmap(r, h) + (e & 1);
    int n = nt * 16 + nl;
    float v = (k < Kd && n < Nd) ? W[(size_t)k * Nd + n] : 0.0f;
    dst[t] = (_Float16)v;
}

// ---------------------------------------------------------------------------
// Embedding: feats = token_emb[tokens] + pos_emb ; copy coords to ping buffer
// ---------------------------------------------------------------------------
__global__ void egnn_embed_kernel(const int* __restrict__ tokens,
                                  const float* __restrict__ temb,
                                  const float* __restrict__ pemb,
                                  const float* __restrict__ coords,
                                  float* __restrict__ feats,
                                  float* __restrict__ coors) {
    int t = blockIdx.x * blockDim.x + threadIdx.x;
    if (t < NODES * FDIM) {
        int node = t >> 6, d = t & 63;
        int i = node % NA;
        int tok = tokens[node];
        feats[t] = temb[tok * FDIM + d] + pemb[i * FDIM + d];
    }
    if (t < NODES * 3) coors[t] = coords[t];
}

// ---------------------------------------------------------------------------
// KNN: one wave32 per atom. Lanes 0..28 hold candidate j; 8 rounds of
// wave-min on (dist2_bits<<32 | j) keys (stable on ties like top_k).
// Also emits rel_k normalized * layer scale (CoorsNorm).
// ---------------------------------------------------------------------------
__global__ void egnn_knn_kernel(const float* __restrict__ coors,
                                const float* __restrict__ cscale_arr, int layer,
                                int* __restrict__ nbr_idx,
                                float* __restrict__ nbr_dist,
                                float* __restrict__ reln) {
    int wavesPerBlock = blockDim.x >> 5;
    int node = blockIdx.x * wavesPerBlock + (threadIdx.x >> 5);
    int lane = threadIdx.x & 31;
    int b = node / NA, i = node % NA;
    const float* cb = coors + (size_t)b * NA * 3;
    float xi = cb[i * 3 + 0], yi = cb[i * 3 + 1], zi = cb[i * 3 + 2];
    float rx = 0.f, ry = 0.f, rz = 0.f, d2 = 0.f;
    unsigned long long key = ~0ull;
    if (lane < NA) {
        rx = xi - cb[lane * 3 + 0];
        ry = yi - cb[lane * 3 + 1];
        rz = zi - cb[lane * 3 + 2];
        d2 = rx * rx + ry * ry + rz * rz;
        key = ((unsigned long long)__float_as_uint(d2) << 32) | (unsigned)lane;
    }
    float scale = cscale_arr[layer];
    for (int kk = 0; kk < KNN; ++kk) {
        unsigned long long m = key;
        for (int off = 16; off; off >>= 1) {
            unsigned hi = __shfl_xor((unsigned)(m >> 32), off, 32);
            unsigned lo = __shfl_xor((unsigned)(m & 0xffffffffu), off, 32);
            unsigned long long o = ((unsigned long long)hi << 32) | lo;
            m = (o < m) ? o : m;
        }
        int jwin = (int)(m & 0xffffffffu);
        if (lane == jwin) {
            int e = node * KNN + kk;
            nbr_idx[e]  = lane;
            nbr_dist[e] = d2;
            float inv = scale / fmaxf(sqrtf(d2), COORS_EPS);
            reln[e * 3 + 0] = rx * inv;
            reln[e * 3 + 1] = ry * inv;
            reln[e * 3 + 2] = rz * inv;
            key = ~0ull;   // remove from next rounds
        }
    }
}

// ---------------------------------------------------------------------------
// Edge MLP kernel: one wave handles 16 edges. WMMA chain:
//   GEMM1 16x160x272 (edge_in @ ew1) -> SiLU
//   GEMM2 16x288x16  (h @ ew2)       -> SiLU -> m_ij (store)
//   GEMM3 16x32x64   (m @ cw1)       -> SiLU
//   dot64 (c1 @ cw2) + clamp -> coor_w (store)
// ---------------------------------------------------------------------------
#define WPB_E 2
__global__ void egnn_edge_kernel(const float* __restrict__ feats,
                                 const int* __restrict__ nbr_idx,
                                 const float* __restrict__ nbr_dist,
                                 const _Float16* __restrict__ pE1,
                                 const _Float16* __restrict__ pE2,
                                 const _Float16* __restrict__ pC1,
                                 const float* __restrict__ eb1,
                                 const float* __restrict__ eb2,
                                 const float* __restrict__ cb1,
                                 const float* __restrict__ cw2,
                                 const float* __restrict__ cb2,
                                 float* __restrict__ m_out,
                                 float* __restrict__ w_out) {
    __shared__ _Float16 sX[WPB_E][16 * 160];
    __shared__ _Float16 sH[WPB_E][16 * 288];
    __shared__ _Float16 sM[WPB_E][16 * 32];
    __shared__ _Float16 sC[WPB_E][16 * 64];

    int wv = threadIdx.x >> 5, lane = threadIdx.x & 31;
    int e0 = (blockIdx.x * WPB_E + wv) * 16;
    _Float16* X  = sX[wv];
    _Float16* H  = sH[wv];
    _Float16* Mh = sM[wv];
    _Float16* C1 = sC[wv];

    // Stage edge inputs (f16) and zero K-pad regions
    for (int t = lane; t < 16 * 160; t += 32) {
        int row = t / 160, c = t - row * 160;
        int e = e0 + row;
        int node = e / KNN;
        int b = node / NA;
        float v = 0.0f;
        if (c < FDIM)            v = feats[(size_t)node * FDIM + c];
        else if (c < 2 * FDIM)   v = feats[((size_t)b * NA + nbr_idx[e]) * FDIM + (c - FDIM)];
        else if (c == 2 * FDIM)  v = nbr_dist[e];
        X[t] = (_Float16)v;
    }
    for (int t = lane; t < 16 * 288; t += 32) H[t]  = (_Float16)0.0f;
    for (int t = lane; t < 16 * 32;  t += 32) Mh[t] = (_Float16)0.0f;
    __syncthreads();

    int idx = lane & 15, h = lane >> 4;

    // Hoist the 5 A-fragments (whole 16x160 LHS) into registers
    v16h a1[E1_KS];
#pragma unroll
    for (int ks = 0; ks < E1_KS; ++ks) a1[ks] = load_a_frag(X, 160, ks * 32, lane);

    // GEMM1: edge_in(129->160) @ ew1 -> 272 cols (258 live)
    for (int nt = 0; nt < E1_NT; ++nt) {
        v8f acc = zero8();
#pragma unroll
        for (int ks = 0; ks < E1_KS; ++ks) {
            v16h bf = *(const v16h*)(pE1 + ((size_t)(ks * E1_NT + nt) * 32 + lane) * 16);
            acc = wmma16(a1[ks], bf, acc);
        }
        int n = nt * 16 + idx;
        // branchless bias (avoid divergent load around WMMA code)
        float bias = eb1[n < EHID ? n : 0];
        bias = (n < EHID) ? bias : 0.0f;
#pragma unroll
        for (int r = 0; r < 8; ++r)
            H[(r + 8 * h) * 288 + n] = (_Float16)silu(acc[r] + bias);
    }
    __syncthreads();

    // GEMM2: h(258->288) @ ew2 -> m_ij (16)
    {
        v8f acc = zero8();
        for (int ks = 0; ks < E2_KS; ++ks) {
            v16h a = load_a_frag(H, 288, ks * 32, lane);
            v16h bf = *(const v16h*)(pE2 + ((size_t)ks * 32 + lane) * 16);
            acc = wmma16(a, bf, acc);
        }
        float bias = eb2[idx];
#pragma unroll
        for (int r = 0; r < 8; ++r) {
            float m = silu(acc[r] + bias);
            Mh[(r + 8 * h) * 32 + idx] = (_Float16)m;
            m_out[(size_t)(e0 + r + 8 * h) * MDIM + idx] = m;
        }
    }
    __syncthreads();

    // GEMM3: m(16->32) @ cw1 -> 64, SiLU
    {
        v16h a = load_a_frag(Mh, 32, 0, lane);
#pragma unroll
        for (int nt = 0; nt < C1_NT; ++nt) {
            v8f acc = zero8();
            v16h bf = *(const v16h*)(pC1 + ((size_t)nt * 32 + lane) * 16);
            acc = wmma16(a, bf, acc);
            int n = nt * 16 + idx;
            float bias = cb1[n];
#pragma unroll
            for (int r = 0; r < 8; ++r)
                C1[(r + 8 * h) * 64 + n] = (_Float16)silu(acc[r] + bias);
        }
    }
    __syncthreads();

    // coor_w = clamp(c1 @ cw2 + cb2, +-2): one lane per edge row
    if (lane < 16) {
        float s = cb2[0];
        for (int d = 0; d < 64; ++d) s += (float)C1[lane * 64 + d] * cw2[d];
        s = fminf(fmaxf(s, -CLAMP_W), CLAMP_W);
        w_out[e0 + lane] = s;
    }
}

// ---------------------------------------------------------------------------
// Node kernel: one wave per 16 nodes. m_i sum + coords update + LayerNorm,
// then WMMA node MLP 16x96x128 -> SiLU -> 16x128x64 + residual.
// ---------------------------------------------------------------------------
#define WPB_N 8
__global__ void egnn_node_kernel(const float* __restrict__ feats_in,
                                 const float* __restrict__ coors_in,
                                 const float* __restrict__ m_ij,
                                 const float* __restrict__ coor_w,
                                 const float* __restrict__ reln,
                                 const float* __restrict__ ln_g,
                                 const float* __restrict__ ln_b,
                                 const _Float16* __restrict__ pN1,
                                 const _Float16* __restrict__ pN2,
                                 const float* __restrict__ nb1,
                                 const float* __restrict__ nb2,
                                 float* __restrict__ feats_out,
                                 float* __restrict__ coors_out) {
    __shared__ _Float16 sX[WPB_N][16 * 96];
    __shared__ _Float16 sH[WPB_N][16 * 128];

    int wv = threadIdx.x >> 5, lane = threadIdx.x & 31;
    int node0 = (blockIdx.x * WPB_N + wv) * 16;
    _Float16* X = sX[wv];
    _Float16* H = sH[wv];

    for (int t = lane; t < 16 * 96; t += 32) X[t] = (_Float16)0.0f;

    // LayerNorm + coords update (lanes 0..15, one node each)
    if (lane < 16) {
        int node = node0 + lane;
        const float* f = feats_in + (size_t)node * FDIM;
        float mu = 0.f;
        for (int d = 0; d < FDIM; ++d) mu += f[d];
        mu *= (1.0f / FDIM);
        float var = 0.f;
        for (int d = 0; d < FDIM; ++d) { float t = f[d] - mu; var += t * t; }
        var *= (1.0f / FDIM);
        float inv = rsqrtf(var + LN_EPS);
        for (int d = 0; d < FDIM; ++d)
            X[lane * 96 + d] = (_Float16)((f[d] - mu) * inv * ln_g[d] + ln_b[d]);

        float cx = coors_in[node * 3 + 0];
        float cy = coors_in[node * 3 + 1];
        float cz = coors_in[node * 3 + 2];
        for (int k = 0; k < KNN; ++k) {
            int e = node * KNN + k;
            float w = coor_w[e];
            cx += w * reln[e * 3 + 0];
            cy += w * reln[e * 3 + 1];
            cz += w * reln[e * 3 + 2];
        }
        coors_out[node * 3 + 0] = cx;
        coors_out[node * 3 + 1] = cy;
        coors_out[node * 3 + 2] = cz;
    }

    // m_i = sum_k m_ij : lane (s = lane%16) handles 8 dims
    {
        int s = lane & 15, db = (lane >> 4) * 8;
        int node = node0 + s;
        for (int d = 0; d < 8; ++d) {
            float acc = 0.f;
            for (int k = 0; k < KNN; ++k)
                acc += m_ij[((size_t)node * KNN + k) * MDIM + db + d];
            X[s * 96 + FDIM + db + d] = (_Float16)acc;
        }
    }
    __syncthreads();

    int idx = lane & 15, h = lane >> 4;

    // Hoist the 3 A-fragments (16x96 LHS)
    v16h an[N1_KS];
#pragma unroll
    for (int ks = 0; ks < N1_KS; ++ks) an[ks] = load_a_frag(X, 96, ks * 32, lane);

    // GEMM1: node_in(80->96) @ nw1 -> 128, SiLU
    for (int nt = 0; nt < N1_NT; ++nt) {
        v8f acc = zero8();
#pragma unroll
        for (int ks = 0; ks < N1_KS; ++ks) {
            v16h bf = *(const v16h*)(pN1 + ((size_t)(ks * N1_NT + nt) * 32 + lane) * 16);
            acc = wmma16(an[ks], bf, acc);
        }
        int n = nt * 16 + idx;
        float bias = nb1[n];
#pragma unroll
        for (int r = 0; r < 8; ++r)
            H[(r + 8 * h) * 128 + n] = (_Float16)silu(acc[r] + bias);
    }
    __syncthreads();

    // Hoist the 4 A-fragments (16x128 hidden)
    v16h ah[N2_KS];
#pragma unroll
    for (int ks = 0; ks < N2_KS; ++ks) ah[ks] = load_a_frag(H, 128, ks * 32, lane);

    // GEMM2: h(128) @ nw2 -> 64, + bias + residual
    for (int nt = 0; nt < N2_NT; ++nt) {
        v8f acc = zero8();
#pragma unroll
        for (int ks = 0; ks < N2_KS; ++ks) {
            v16h bf = *(const v16h*)(pN2 + ((size_t)(ks * N2_NT + nt) * 32 + lane) * 16);
            acc = wmma16(ah[ks], bf, acc);
        }
        int n = nt * 16 + idx;
        float bias = nb2[n];
#pragma unroll
        for (int r = 0; r < 8; ++r) {
            int row = node0 + r + 8 * h;
            feats_out[(size_t)row * FDIM + n] =
                acc[r] + bias + feats_in[(size_t)row * FDIM + n];
        }
    }
}

// ---------------------------------------------------------------------------
// Readout: wave per molecule; masked mean (mask all-true -> /29), 64->64 SiLU,
// 64->1 with wave reduction.
// ---------------------------------------------------------------------------
#define WPB_R 8
__global__ void egnn_readout_kernel(const float* __restrict__ feats,
                                    const float* __restrict__ rw1,
                                    const float* __restrict__ rb1,
                                    const float* __restrict__ rw2,
                                    const float* __restrict__ rb2,
                                    float* __restrict__ out) {
    __shared__ float sMol[WPB_R][64];
    int wv = threadIdx.x >> 5, lane = threadIdx.x & 31;
    int b = blockIdx.x * WPB_R + wv;
    const float* f = feats + (size_t)b * NA * FDIM;
    int d0 = lane * 2;
    float m0 = 0.f, m1 = 0.f;
    for (int n = 0; n < NA; ++n) {
        m0 += f[n * FDIM + d0];
        m1 += f[n * FDIM + d0 + 1];
    }
    sMol[wv][d0]     = m0 * (1.0f / NA);
    sMol[wv][d0 + 1] = m1 * (1.0f / NA);
    __syncthreads();
    float h0 = rb1[d0], h1 = rb1[d0 + 1];
    for (int d = 0; d < 64; ++d) {
        float mv = sMol[wv][d];
        h0 += mv * rw1[d * 64 + d0];
        h1 += mv * rw1[d * 64 + d0 + 1];
    }
    h0 = silu(h0);
    h1 = silu(h1);
    float p = h0 * rw2[d0] + h1 * rw2[d0 + 1];
    for (int off = 16; off; off >>= 1) p += __shfl_xor(p, off, 32);
    if (lane == 0) out[b] = p + rb2[0];
}

// ---------------------------------------------------------------------------
// Host orchestration
// ---------------------------------------------------------------------------
extern "C" void kernel_launch(void* const* d_in, const int* in_sizes, int n_in,
                              void* d_out, int out_size, void* d_ws, size_t ws_size,
                              hipStream_t stream) {
    (void)in_sizes; (void)n_in; (void)out_size; (void)ws_size;
    const int*   tokens = (const int*)d_in[0];
    const float* coords = (const float*)d_in[1];
    // d_in[2] = mask (all true in this problem) - intentionally unused
    const float* temb = (const float*)d_in[3];
    const float* pemb = (const float*)d_in[4];
    const float* ew1  = (const float*)d_in[5];
    const float* eb1  = (const float*)d_in[6];
    const float* ew2  = (const float*)d_in[7];
    const float* eb2  = (const float*)d_in[8];
    const float* cw1  = (const float*)d_in[9];
    const float* cb1  = (const float*)d_in[10];
    const float* cw2  = (const float*)d_in[11];
    const float* cb2  = (const float*)d_in[12];
    const float* cscale = (const float*)d_in[13];
    const float* ln_g = (const float*)d_in[14];
    const float* ln_b = (const float*)d_in[15];
    const float* nw1  = (const float*)d_in[16];
    const float* nb1  = (const float*)d_in[17];
    const float* nw2  = (const float*)d_in[18];
    const float* nb2  = (const float*)d_in[19];
    const float* rw1  = (const float*)d_in[20];
    const float* rb1  = (const float*)d_in[21];
    const float* rw2  = (const float*)d_in[22];
    const float* rb2  = (const float*)d_in[23];

    // Workspace carve-out (~38 MB total)
    char* p = (char*)d_ws;
    auto alloc = [&](size_t bytes) -> void* {
        void* r = (void*)p;
        p += (bytes + 255) & ~(size_t)255;
        return r;
    };
    float* feats0 = (float*)alloc((size_t)NODES * FDIM * 4);
    float* feats1 = (float*)alloc((size_t)NODES * FDIM * 4);
    float* coors0 = (float*)alloc((size_t)NODES * 3 * 4);
    float* coors1 = (float*)alloc((size_t)NODES * 3 * 4);
    int*   nidx   = (int*)alloc((size_t)EDGES * 4);
    float* ndist  = (float*)alloc((size_t)EDGES * 4);
    float* reln   = (float*)alloc((size_t)EDGES * 3 * 4);
    float* m_ij   = (float*)alloc((size_t)EDGES * MDIM * 4);
    float* corw   = (float*)alloc((size_t)EDGES * 4);
    _Float16* pE1 = (_Float16*)alloc((size_t)DEPTH * PE1_HALFS * 2);
    _Float16* pE2 = (_Float16*)alloc((size_t)DEPTH * PE2_HALFS * 2);
    _Float16* pC1 = (_Float16*)alloc((size_t)DEPTH * PC1_HALFS * 2);
    _Float16* pN1 = (_Float16*)alloc((size_t)DEPTH * PN1_HALFS * 2);
    _Float16* pN2 = (_Float16*)alloc((size_t)DEPTH * PN2_HALFS * 2);

    // Pack weights into WMMA B-fragment layout (once per launch)
    for (int l = 0; l < DEPTH; ++l) {
        egnn_pack_kernel<<<(PE1_HALFS + 255) / 256, 256, 0, stream>>>(
            ew1 + (size_t)l * EDGE_IN * EHID, EDGE_IN, EHID, E1_KS, E1_NT,
            pE1 + (size_t)l * PE1_HALFS);
        egnn_pack_kernel<<<(PE2_HALFS + 255) / 256, 256, 0, stream>>>(
            ew2 + (size_t)l * EHID * MDIM, EHID, MDIM, E2_KS, 1,
            pE2 + (size_t)l * PE2_HALFS);
        egnn_pack_kernel<<<(PC1_HALFS + 255) / 256, 256, 0, stream>>>(
            cw1 + (size_t)l * MDIM * 64, MDIM, 64, 1, C1_NT,
            pC1 + (size_t)l * PC1_HALFS);
        egnn_pack_kernel<<<(PN1_HALFS + 255) / 256, 256, 0, stream>>>(
            nw1 + (size_t)l * NIN * NHID, NIN, NHID, N1_KS, N1_NT,
            pN1 + (size_t)l * PN1_HALFS);
        egnn_pack_kernel<<<(PN2_HALFS + 255) / 256, 256, 0, stream>>>(
            nw2 + (size_t)l * NHID * FDIM, NHID, FDIM, N2_KS, N2_NT,
            pN2 + (size_t)l * PN2_HALFS);
    }

    // Embedding + coord copy
    egnn_embed_kernel<<<(NODES * FDIM + 255) / 256, 256, 0, stream>>>(
        tokens, temb, pemb, coords, feats0, coors0);

    float* fcur = feats0; float* fnxt = feats1;
    float* ccur = coors0; float* cnxt = coors1;
    for (int l = 0; l < DEPTH; ++l) {
        // KNN: 8 waves/block -> 8 nodes/block
        egnn_knn_kernel<<<NODES / 8, 256, 0, stream>>>(
            ccur, cscale, l, nidx, ndist, reln);
        // Edge MLP: 2 waves/block, 16 edges/wave
        egnn_edge_kernel<<<EDGES / (WPB_E * 16), WPB_E * 32, 0, stream>>>(
            fcur, nidx, ndist,
            pE1 + (size_t)l * PE1_HALFS, pE2 + (size_t)l * PE2_HALFS,
            pC1 + (size_t)l * PC1_HALFS,
            eb1 + (size_t)l * EHID, eb2 + (size_t)l * MDIM,
            cb1 + (size_t)l * 64, cw2 + (size_t)l * 64, cb2 + l,
            m_ij, corw);
        // Node MLP: 8 waves/block, 16 nodes/wave
        egnn_node_kernel<<<NODES / (WPB_N * 16), WPB_N * 32, 0, stream>>>(
            fcur, ccur, m_ij, corw, reln,
            ln_g + (size_t)l * FDIM, ln_b + (size_t)l * FDIM,
            pN1 + (size_t)l * PN1_HALFS, pN2 + (size_t)l * PN2_HALFS,
            nb1 + (size_t)l * NHID, nb2 + (size_t)l * FDIM,
            fnxt, cnxt);
        float* t;
        t = fcur; fcur = fnxt; fnxt = t;
        t = ccur; ccur = cnxt; cnxt = t;
    }

    egnn_readout_kernel<<<BATCH / WPB_R, WPB_R * 32, 0, stream>>>(
        fcur, rw1, rb1, rw2, rb2, (float*)d_out);
}
